// NodeInference_2138893713790
// MI455X (gfx1250) — matrix operand order repaired and checked by reading, not compiled, not run on hardware
//
#include <hip/hip_runtime.h>
#include <hip/hip_bf16.h>

// ---------------------------------------------------------------------------
// Types for WMMA fragments (gfx1250, wave32)
// ---------------------------------------------------------------------------
typedef __attribute__((ext_vector_type(16))) _Float16 v16h;
typedef __attribute__((ext_vector_type(8)))  float    v8f;

#define HEADS1 2
#define HID    128
#define F1     256   /* HEADS1*HID, also IN features */
#define KDIM   8
#define MDIM   128
#define GCOLS  1024  /* KDIM*MDIM */
#define NKC    (F1 / 32)   /* k-chunks of 32 per GEMM (8) */

// ---------------------------------------------------------------------------
// WMMA compute: D = A(16x32 f16) x B(32x16 f16) + C(16x16 f32)
// ---------------------------------------------------------------------------
__device__ __forceinline__ v8f wmma_f16(v16h a, v16h b, v8f c) {
    return __builtin_amdgcn_wmma_f32_16x16x32_f16(
        /*neg_a=*/false, a, /*neg_b=*/false, b,
        /*c_mod=*/(short)0, c, /*reuse_a=*/false, /*reuse_b=*/false);
}

// C/D: 16x16 f32. VGPR r: lanes 0-15 -> (M=r, N=lane); lanes 16-31 -> (M=r+8, N=lane-16)
__device__ __forceinline__ void store_frag_c(float* C, int ldc, int lane, v8f acc) {
    float* p = C + (size_t)((lane >> 4) << 3) * ldc + (lane & 15);
#pragma unroll
    for (int r = 0; r < 8; ++r) p[r * ldc] = acc[r];
}

// ---------------------------------------------------------------------------
// Fragment pre-packing: store f16 fragments in a per-lane register
// layout (k = 16*(lane/16) + elem, same for A and B) so the GEMM hot loop is pure b128 loads + wmma.
// Packed unit = 32 lanes x 16 halfs (32B/lane, contiguous, coalesced).
// ---------------------------------------------------------------------------

// A-side: X row-major f32 [N, Kdim] -> fragments per (rowTile, kChunk).
// Lane layout: lanes 0-15: M=lane, K = +0..15 (elems 0-15)
//              lanes 16-31: M=lane-16, K = +16..31
__global__ void pack_a(const float* __restrict__ X, _Float16* __restrict__ P,
                       int N, int Kdim) {
    const int lane = threadIdx.x & 31;
    const int kc   = threadIdx.x >> 5;     // 8 waves == 8 k-chunks for Kdim=256
    const int rt   = blockIdx.x;
    const int nkc  = Kdim >> 5;
    if (kc >= nkc) return;
    int row = rt * 16 + (lane & 15);
    if (row >= N) row = N - 1;             // duplicate pad row (never stored)
    const float* src = X + (size_t)row * Kdim + kc * 32 + ((lane >> 4) << 4);
    v16h v;
#pragma unroll
    for (int j = 0; j < 16; ++j) v[j] = (_Float16)src[j];
    size_t unit = ((size_t)rt * nkc + kc) * 32 + lane;
    *(v16h*)(P + unit * 16) = v;
}

// B-side from W[Fout,Kdim] row-major (computing X @ W^T): B[k,n] = W[colBase+n, k].
// Lane layout: lanes 0-15 hold K=+0..15 for col n=lane; lanes 16-31 K=+16..31, n=lane-16.
__global__ void pack_b_wt(const float* __restrict__ W, _Float16* __restrict__ P,
                          int Fout, int Kdim) {
    const int lane = threadIdx.x & 31;
    const int kc   = threadIdx.x >> 5;
    const int ct   = blockIdx.x;
    const int nkc  = Kdim >> 5;
    if (kc >= nkc) return;
    int col = ct * 16 + (lane & 15);
    const float* src = W + (size_t)col * Kdim + kc * 32 + ((lane >> 4) << 4);
    v16h v;
#pragma unroll
    for (int j = 0; j < 16; ++j) v[j] = (_Float16)src[j];
    size_t unit = ((size_t)ct * nkc + kc) * 32 + lane;
    *(v16h*)(P + unit * 16) = v;
}

// B-side from G[Kdim,Ncols] row-major (computing H @ G): B[k,n] = G[k, colBase+n].
__global__ void pack_b_rm(const float* __restrict__ G, _Float16* __restrict__ P,
                          int Kdim, int Ncols) {
    const int lane = threadIdx.x & 31;
    const int kc   = threadIdx.x >> 5;
    const int ct   = blockIdx.x;
    const int nkc  = Kdim >> 5;
    if (kc >= nkc) return;
    int col   = ct * 16 + (lane & 15);
    int kbase = kc * 32 + ((lane >> 4) << 4);
    v16h v;
#pragma unroll
    for (int j = 0; j < 16; ++j) v[j] = (_Float16)G[(size_t)(kbase + j) * Ncols + col];
    size_t unit = ((size_t)ct * nkc + kc) * 32 + lane;
    *(v16h*)(P + unit * 16) = v;
}

// ---------------------------------------------------------------------------
// GEMM on pre-packed fragments: H[N,Fout] = X @ W^T.
// grid.x = row tiles, grid.y*8 waves = column tiles. Hot loop: loads + wmma only.
// ---------------------------------------------------------------------------
__global__ void gemm_wmma_packed(const _Float16* __restrict__ Apk,
                                 const _Float16* __restrict__ Bpk,
                                 float* __restrict__ H, int N, int nkc, int Fout) {
    const int lane = threadIdx.x & 31;
    const int wave = threadIdx.x >> 5;
    const int rt = blockIdx.x;
    const int ct = blockIdx.y * 8 + wave;
    if (rt * 16 >= N || ct * 16 >= Fout) return;   // uniform: EXEC all-1s at wmma

    const v16h* ap = (const v16h*)Apk + (size_t)rt * nkc * 32 + lane;
    const v16h* bp = (const v16h*)Bpk + (size_t)ct * nkc * 32 + lane;
    v8f acc = {};
#pragma unroll
    for (int kc = 0; kc < NKC; ++kc)
        acc = wmma_f16(ap[kc * 32], bp[kc * 32], acc);
    store_frag_c(H + (size_t)rt * 16 * Fout + ct * 16, Fout, lane, acc);
}

// ---------------------------------------------------------------------------
// Per-node attention logits, wave per node (coalesced):
// alpha[n,h] = sum_d H[n,h*dim+d]*a[h,d].  heads <= 2.
// ---------------------------------------------------------------------------
__global__ void alpha_kernel(const float* __restrict__ H,
                             const float* __restrict__ a_s, const float* __restrict__ a_d,
                             float* __restrict__ os, float* __restrict__ od,
                             int N, int heads, int dim) {
    const int lane = threadIdx.x & 31;
    const int n = blockIdx.x * (blockDim.x >> 5) + (threadIdx.x >> 5);
    if (n >= N) return;
    const int F = heads * dim;
    float ss0 = 0.f, sd0 = 0.f, ss1 = 0.f, sd1 = 0.f;
    for (int f = lane; f < F; f += 32) {
        float v  = H[(size_t)n * F + f];
        float vs = v * a_s[f];
        float vd = v * a_d[f];
        if (f < dim) { ss0 += vs; sd0 += vd; } else { ss1 += vs; sd1 += vd; }
    }
#pragma unroll
    for (int off = 16; off > 0; off >>= 1) {
        ss0 += __shfl_xor(ss0, off, 32); sd0 += __shfl_xor(sd0, off, 32);
        ss1 += __shfl_xor(ss1, off, 32); sd1 += __shfl_xor(sd1, off, 32);
    }
    if (lane == 0) {
        os[(size_t)n * heads] = ss0;
        od[(size_t)n * heads] = sd0;
        if (heads > 1) { os[(size_t)n * heads + 1] = ss1; od[(size_t)n * heads + 1] = sd1; }
    }
}

// ---------------------------------------------------------------------------
// Order-preserving float<->uint encoding for atomicMax-based segment max
// ---------------------------------------------------------------------------
__device__ __forceinline__ unsigned fenc(float f) {
    unsigned u = __float_as_uint(f);
    return (u >> 31) ? ~u : (u | 0x80000000u);
}
__device__ __forceinline__ float fdec(unsigned e) {
    unsigned u = (e >> 31) ? (e & 0x7fffffffu) : ~e;
    return __uint_as_float(u);
}

__device__ __forceinline__ void edge_endpoints(const long long* ei, long long e,
                                               long long E, long long& s, long long& d) {
    if (e < E) { s = ei[e]; d = ei[E + e]; }
    else       { s = d = e - E; }            // implicit self-loop
}

// Pass 1: segment max (encoded atomicMax)
__global__ void edge_max(const long long* __restrict__ ei,
                         const float* __restrict__ as, const float* __restrict__ ad,
                         unsigned* __restrict__ m, long long E, int N, int heads, float slope) {
    long long idx = (long long)blockIdx.x * blockDim.x + threadIdx.x;
    long long total = (E + N) * heads;
    if (idx >= total) return;
    int h = (int)(idx % heads);
    long long e = idx / heads, s, d;
    edge_endpoints(ei, e, E, s, d);
    float v = as[s * heads + h] + ad[d * heads + h];
    v = v > 0.f ? v : v * slope;
    atomicMax(&m[d * heads + h], fenc(v));
}

// Pass 2: w = exp(e - max); segment sum
__global__ void edge_expsum(const long long* __restrict__ ei,
                            const float* __restrict__ as, const float* __restrict__ ad,
                            const unsigned* __restrict__ m, float* __restrict__ we,
                            float* __restrict__ ssum, long long E, int N, int heads, float slope) {
    long long idx = (long long)blockIdx.x * blockDim.x + threadIdx.x;
    long long total = (E + N) * heads;
    if (idx >= total) return;
    int h = (int)(idx % heads);
    long long e = idx / heads, s, d;
    edge_endpoints(ei, e, E, s, d);
    float v = as[s * heads + h] + ad[d * heads + h];
    v = v > 0.f ? v : v * slope;
    float w = __expf(v - fdec(m[d * heads + h]));
    we[idx] = w;
    atomicAdd(&ssum[d * heads + h], w);
}

// Pass 3: Hout[dst] += Hin[src] * (w / s[dst]);  one wave per edge, lane per feature
__global__ void edge_scatter(const long long* __restrict__ ei,
                             const float* __restrict__ we, const float* __restrict__ ssum,
                             const float* __restrict__ Hin, float* __restrict__ Hout,
                             long long E, int N, int F, int heads, int dim) {
    int lane = threadIdx.x & 31;
    long long e = (long long)blockIdx.x * (blockDim.x >> 5) + (threadIdx.x >> 5);
    if (e >= E + N) return;
    long long s, d;
    edge_endpoints(ei, e, E, s, d);
    for (int f = lane; f < F; f += 32) {
        int h = f / dim;
        float coef = we[e * heads + h] / ssum[d * heads + h];
        atomicAdd(&Hout[d * (size_t)F + f], Hin[s * (size_t)F + f] * coef);
    }
}

// Bias + optional leaky-relu
__global__ void bias_act(float* __restrict__ H, const float* __restrict__ b,
                         int N, int F, float slope, int act) {
    long long i = (long long)blockIdx.x * blockDim.x + threadIdx.x;
    if (i >= (long long)N * F) return;
    float v = H[i] + b[i % F];
    if (act) v = v > 0.f ? v : v * slope;
    H[i] = v;
}

// ---------------------------------------------------------------------------
// Fused final: t = H2[N,256] @ G[256,1024]; cosine vs mu per k without
// materializing t. Wave w owns k=w (128 cols = 8 WMMA tiles of f32 accs).
// A and G pre-packed.
// ---------------------------------------------------------------------------
__global__ void gemm_cos_wmma(const _Float16* __restrict__ Apk,
                              const _Float16* __restrict__ Gpk,
                              const float* __restrict__ MU, float* __restrict__ OUT, int N) {
    const int lane = threadIdx.x & 31;
    const int k = threadIdx.x >> 5;          // wave id == output k (0..7)
    const int rt = blockIdx.x;
    if (rt * 16 >= N) return;

    const v16h* ap = (const v16h*)Apk + (size_t)rt * NKC * 32 + lane;
    const v16h* gp = (const v16h*)Gpk;

    v8f acc[8];
#pragma unroll
    for (int t = 0; t < 8; ++t) acc[t] = (v8f){};

#pragma unroll
    for (int kc = 0; kc < NKC; ++kc) {
        v16h a = ap[kc * 32];
#pragma unroll
        for (int t = 0; t < 8; ++t) {
            size_t ct = (size_t)(k * 8 + t);
            v16h b = gp[(ct * NKC + kc) * 32 + lane];
            acc[t] = wmma_f16(a, b, acc[t]);
        }
    }

    // ||mu[k]|| : full-wave reduce
    float musq = 0.f;
    for (int j = lane; j < MDIM; j += 32) { float mv = MU[k * MDIM + j]; musq += mv * mv; }
#pragma unroll
    for (int off = 16; off > 0; off >>= 1) musq += __shfl_xor(musq, off, 32);
    float munorm = __builtin_sqrtf(musq);

    // Per-lane partials per row-VGPR r (C layout: M = r + 8*(lane>=16), N = lane&15)
    float num[8], sq[8];
#pragma unroll
    for (int r = 0; r < 8; ++r) { num[r] = 0.f; sq[r] = 0.f; }
    int nc = lane & 15;
#pragma unroll
    for (int t = 0; t < 8; ++t) {
        float muv = MU[k * MDIM + t * 16 + nc];
#pragma unroll
        for (int r = 0; r < 8; ++r) {
            float v = acc[t][r];
            num[r] += v * muv;
            sq[r]  += v * v;
        }
    }
    // Reduce across the 16 lanes sharing each row (xor 1,2,4,8 stays within halves)
#pragma unroll
    for (int off = 1; off < 16; off <<= 1) {
#pragma unroll
        for (int r = 0; r < 8; ++r) {
            num[r] += __shfl_xor(num[r], off, 32);
            sq[r]  += __shfl_xor(sq[r],  off, 32);
        }
    }
    if ((lane & 15) == 0) {
        int mbase = (lane >> 4) * 8;
#pragma unroll
        for (int r = 0; r < 8; ++r) {
            int node = rt * 16 + mbase + r;
            if (node < N) {
                float denom = fmaxf(__builtin_sqrtf(sq[r]) * munorm, 1e-8f);
                OUT[(size_t)node * KDIM + k] = num[r] / denom;
            }
        }
    }
}

// ---------------------------------------------------------------------------
// Host launcher
// ---------------------------------------------------------------------------
extern "C" void kernel_launch(void* const* d_in, const int* in_sizes, int n_in,
                              void* d_out, int out_size, void* d_ws, size_t ws_size,
                              hipStream_t stream) {
    const float*     x    = (const float*)d_in[0];
    const long long* ei   = (const long long*)d_in[1];
    const float*     W1   = (const float*)d_in[2];
    const float*     a1sw = (const float*)d_in[3];
    const float*     a1dw = (const float*)d_in[4];
    const float*     b1   = (const float*)d_in[5];
    const float*     W2   = (const float*)d_in[6];
    const float*     a2sw = (const float*)d_in[7];
    const float*     a2dw = (const float*)d_in[8];
    const float*     b2   = (const float*)d_in[9];
    const float*     g    = (const float*)d_in[10];
    const float*     mu   = (const float*)d_in[11];
    float*           out  = (float*)d_out;

    const int       N    = in_sizes[0] / F1;
    const long long E    = in_sizes[1] / 2;
    const long long Etot = E + (long long)N;
    const int rowTiles   = (N + 15) / 16;

    // ---- Workspace carve (32B-aligned packed f16 first, then f32 buffers) ----
    _Float16* hp = (_Float16*)d_ws;
    _Float16* xp  = hp; hp += (size_t)rowTiles * 16 * F1;  // packed A fragments
    _Float16* W1p = hp; hp += (size_t)F1 * F1;
    _Float16* W2p = hp; hp += (size_t)F1 * F1;
    _Float16* gpk = hp; hp += (size_t)F1 * GCOLS;

    float* p  = (float*)hp;
    float* hA = p; p += (size_t)N * F1;      // layer linear outputs (reused)
    float* hB = p; p += (size_t)N * F1;      // aggregation targets (reused)
    float* a1s = p; p += (size_t)N * HEADS1;
    float* a1d = p; p += (size_t)N * HEADS1;
    float* a2s = p; p += (size_t)N;
    float* a2d = p; p += (size_t)N;
    unsigned* m1 = (unsigned*)p; p += (size_t)N * HEADS1;
    float* s1 = p; p += (size_t)N * HEADS1;
    unsigned* m2 = (unsigned*)p; p += (size_t)N;
    float* s2 = p; p += (size_t)N;
    float* w1e = p; p += (size_t)Etot * HEADS1;
    float* w2e = p; p += (size_t)Etot;

    // ---- Weight packing (tiny, once per launch) ----
    pack_b_wt<<<F1 / 16,    256, 0, stream>>>(W1, W1p, F1, F1);
    pack_b_wt<<<F1 / 16,    256, 0, stream>>>(W2, W2p, F1, F1);
    pack_b_rm<<<GCOLS / 16, 256, 0, stream>>>(g,  gpk, F1, GCOLS);

    // ---- Layer 1 ----
    hipMemsetAsync(m1, 0, (size_t)N * HEADS1 * sizeof(unsigned), stream);
    hipMemsetAsync(s1, 0, (size_t)N * HEADS1 * sizeof(float), stream);
    hipMemsetAsync(hB, 0, (size_t)N * F1 * sizeof(float), stream);

    pack_a<<<rowTiles, 256, 0, stream>>>(x, xp, N, F1);
    gemm_wmma_packed<<<dim3(rowTiles, F1 / 128), 256, 0, stream>>>(xp, W1p, hA, N, NKC, F1);
    alpha_kernel<<<(N + 7) / 8, 256, 0, stream>>>(hA, a1sw, a1dw, a1s, a1d, N, HEADS1, HID);

    {
        long long tot = Etot * HEADS1;
        int grid = (int)((tot + 255) / 256);
        edge_max   <<<grid, 256, 0, stream>>>(ei, a1s, a1d, m1, E, N, HEADS1, 0.2f);
        edge_expsum<<<grid, 256, 0, stream>>>(ei, a1s, a1d, m1, w1e, s1, E, N, HEADS1, 0.2f);
    }
    edge_scatter<<<(int)((Etot + 7) / 8), 256, 0, stream>>>(ei, w1e, s1, hA, hB, E, N, F1, HEADS1, HID);
    bias_act<<<(int)(((long long)N * F1 + 255) / 256), 256, 0, stream>>>(hB, b1, N, F1, 0.01f, 1);

    // ---- Layer 2 (1 head, 256 dims) ----
    pack_a<<<rowTiles, 256, 0, stream>>>(hB, xp, N, F1);
    gemm_wmma_packed<<<dim3(rowTiles, F1 / 128), 256, 0, stream>>>(xp, W2p, hA, N, NKC, F1);
    alpha_kernel<<<(N + 7) / 8, 256, 0, stream>>>(hA, a2sw, a2dw, a2s, a2d, N, 1, F1);

    hipMemsetAsync(m2, 0, (size_t)N * sizeof(unsigned), stream);
    hipMemsetAsync(s2, 0, (size_t)N * sizeof(float), stream);
    hipMemsetAsync(hB, 0, (size_t)N * F1 * sizeof(float), stream);  // after GEMM2 is queued

    {
        int grid = (int)((Etot + 255) / 256);
        edge_max   <<<grid, 256, 0, stream>>>(ei, a2s, a2d, m2, E, N, 1, 0.2f);
        edge_expsum<<<grid, 256, 0, stream>>>(ei, a2s, a2d, m2, w2e, s2, E, N, 1, 0.2f);
    }
    edge_scatter<<<(int)((Etot + 7) / 8), 256, 0, stream>>>(ei, w2e, s2, hA, hB, E, N, F1, 1, F1);
    bias_act<<<(int)(((long long)N * F1 + 255) / 256), 256, 0, stream>>>(hB, b2, N, F1, 0.f, 0);

    // ---- Fused projection + cosine (A and G pre-packed) ----
    pack_a<<<rowTiles, 256, 0, stream>>>(hB, xp, N, F1);
    gemm_cos_wmma<<<rowTiles, 256, 0, stream>>>(xp, gpk, mu, out, N);
}
